// PointNet2BinSeg_62319975465096
// MI455X (gfx1250) — matrix operand order repaired and checked by reading, not compiled, not run on hardware
//
#include <hip/hip_runtime.h>
#include <hip/hip_bf16.h>
#include <stdint.h>

// ---------------------------------------------------------------------------
// PointNet++ binary segmentation forward for MI455X (gfx1250, wave32, WMMA).
// 1x1-conv layers: v_wmma_f32_16x16x32_f16, weight panel staged to LDS by the
// Tensor Data Mover (tensor_load_to_lds + s_wait_tensorcnt), consumed with
// ds_load_b128 by 4 waves / block, NT=2..4 output tiles per wave (A reuse).
// ---------------------------------------------------------------------------

#define BATCH 8
#define NPTS  4096

typedef __attribute__((ext_vector_type(16))) _Float16 v16h;
typedef __attribute__((ext_vector_type(8)))  _Float16 v8h;
typedef __attribute__((ext_vector_type(8)))  float    v8f;
typedef __attribute__((ext_vector_type(4)))  unsigned int u32x4;
typedef __attribute__((ext_vector_type(8)))  int      i32x8;
typedef __attribute__((ext_vector_type(4)))  int      i32x4;

static inline int rup(int x, int a) { return (x + a - 1) / a * a; }

// ---------------------------------------------------------------------------
// WMMA GEMM: Y[M,Nc] = Xh[M,Kp] * Wh[Nc,Kp]^T + bias[Nc]
// Block = 128 threads (4 waves) covering 4 consecutive M tiles on one N strip
// of NT tiles. Weight panel (NT*16 rows x Kp halves, contiguous in Wh) is
// DMA'd to LDS via TDM, then all waves read B fragments from LDS.
// M % 64 == 0, Nc % (16*NT) == 0, Kp % 32 == 0 (enforced by caller).
// ---------------------------------------------------------------------------
template <int NT>
__global__ void __launch_bounds__(128)
wmma_gemm_panel(const _Float16* __restrict__ Xh, const _Float16* __restrict__ Wh,
                const float* __restrict__ bias, float* __restrict__ Y,
                int M, int Nc, int Kp)
{
    extern __shared__ _Float16 panel[];           // NT*16*Kp halves
    const int lane = threadIdx.x & 31;
    const int wave = threadIdx.x >> 5;
    const int tm   = blockIdx.x * 4 + wave;       // M tile (exact, no guard)
    const int colBase = blockIdx.y * (NT * 16);   // N strip base column
    const int panelElems = NT * 16 * Kp;

    // ---- stage weight panel into LDS ----
#if __has_builtin(__builtin_amdgcn_tensor_load_to_lds)
    if (threadIdx.x < 32) {                       // wave-uniform branch
        uint64_t ga = (uint64_t)(uintptr_t)(Wh + (size_t)colBase * Kp);
        uint32_t lds_off = (uint32_t)(uintptr_t)panel;   // low 32b = LDS offset
        uint32_t L = (uint32_t)panelElems;        // elements (2B each), <= 49152
        u32x4 g0;
        g0[0] = 1u;                               // count=1, user descriptor
        g0[1] = lds_off;                          // lds_addr
        g0[2] = (uint32_t)ga;                     // global_addr[31:0]
        g0[3] = (uint32_t)((ga >> 32) & 0x1FFFFFFu) | (2u << 30); // ga[56:32] | type=2
        i32x8 g1;
        g1[0] = (int)(1u << 16);                  // data_size=1 (2-byte elements)
        g1[1] = (int)((L & 0xFFFFu) << 16);       // tensor_dim0[15:0]
        g1[2] = (int)((L >> 16) | (1u << 16));    // tensor_dim0[31:16] | tensor_dim1=1
        g1[3] = (int)((L & 0xFFFFu) << 16);       // tile_dim0 = L
        g1[4] = 1;                                // tile_dim1=1, tile_dim2=0
        g1[5] = (int)L;                           // tensor_dim0_stride[31:0]
        g1[6] = 0;
        g1[7] = 0;
        i32x4 g2 = {0, 0, 0, 0}, g3 = {0, 0, 0, 0};
        i32x8 g4 = {0, 0, 0, 0, 0, 0, 0, 0};
        __builtin_amdgcn_tensor_load_to_lds(g0, g1, g2, g3, g4, 0);
        __builtin_amdgcn_s_wait_tensorcnt(0);
    }
#else
    for (int i = threadIdx.x; i < panelElems / 8; i += 128)
        *(v8h*)(panel + (size_t)i * 8) =
            *(const v8h*)(Wh + (size_t)colBase * Kp + (size_t)i * 8);
#endif
    __syncthreads();

    const int h16 = lane >> 4;     // 0 or 1
    const int l15 = lane & 15;

    // A: row = tm*16 + l15; lane-half selects K sub-blocks [k+8h,k+8h+8) and +16
    const _Float16* aRow = Xh + (size_t)(tm * 16 + l15) * Kp + h16 * 8;
    // B in LDS: lane holds column N = l15 (within tile), K sub-blocks [k+8h,k+8h+8) and +16
    const _Float16* bLds = panel + (size_t)l15 * Kp + h16 * 8;

    v8f acc[NT];
#pragma unroll
    for (int t = 0; t < NT; ++t) acc[t] = (v8f){};

    for (int k = 0; k < Kp; k += 32) {
        v8h a0 = *(const v8h*)(aRow + k);
        v8h a1 = *(const v8h*)(aRow + k + 16);
        v16h a;
#pragma unroll
        for (int i = 0; i < 8; ++i) { a[i] = a0[i]; a[i + 8] = a1[i]; }
#pragma unroll
        for (int t = 0; t < NT; ++t) {
            const _Float16* bp = bLds + (size_t)t * 16 * Kp + k;
            v8h b0 = *(const v8h*)bp;
            v8h b1 = *(const v8h*)(bp + 16);
            v16h b;
#pragma unroll
            for (int i = 0; i < 8; ++i) { b[i] = b0[i]; b[i + 8] = b1[i]; }
            acc[t] = __builtin_amdgcn_wmma_f32_16x16x32_f16(
                false, a, false, b, (short)0, acc[t], false, false);
        }
    }

    const int mBase = tm * 16 + h16 * 8;
#pragma unroll
    for (int t = 0; t < NT; ++t) {
        const int col  = colBase + t * 16 + l15;
        const float bv = bias[col];
#pragma unroll
        for (int r = 0; r < 8; ++r)
            Y[(size_t)(mBase + r) * Nc + col] = acc[t][r] + bv;
    }
}

// ---------------------------------------------------------------------------
// Train-mode BN statistics: one block per channel, LDS tree reduction.
// ---------------------------------------------------------------------------
__global__ void __launch_bounds__(256)
bn_stats(const float* __restrict__ Y, int M, int Nc,
         const float* __restrict__ g, const float* __restrict__ beta,
         float* __restrict__ scale, float* __restrict__ shift)
{
    __shared__ float ssum[256], ssq[256];
    const int c = blockIdx.x;
    float s = 0.f, s2 = 0.f;
    for (int m = threadIdx.x; m < M; m += 256) {
        float y = Y[(size_t)m * Nc + c];
        s += y; s2 += y * y;
    }
    ssum[threadIdx.x] = s; ssq[threadIdx.x] = s2;
    __syncthreads();
    for (int st = 128; st > 0; st >>= 1) {
        if (threadIdx.x < st) {
            ssum[threadIdx.x] += ssum[threadIdx.x + st];
            ssq[threadIdx.x]  += ssq[threadIdx.x + st];
        }
        __syncthreads();
    }
    if (threadIdx.x == 0) {
        float mean = ssum[0] / (float)M;
        float var  = ssq[0] / (float)M - mean * mean;
        float sc   = g[c] * rsqrtf(var + 1e-5f);
        scale[c] = sc;
        shift[c] = beta[c] - mean * sc;
    }
}

// BN apply + ReLU, in-place on f32 Y, fused f16 conversion for next layer.
__global__ void __launch_bounds__(256)
bn_apply_relu(float* __restrict__ Y, _Float16* __restrict__ Xn,
              const float* __restrict__ scale, const float* __restrict__ shift,
              size_t total, int Nc)
{
    size_t i = (size_t)blockIdx.x * blockDim.x + threadIdx.x;
    if (i >= total) return;
    int c = (int)(i % Nc);
    float y = fmaxf(Y[i] * scale[c] + shift[c], 0.f);
    Y[i] = y;
    if (Xn) Xn[i] = (_Float16)y;
}

// Weight f32 -> f16 with K zero-padding.
__global__ void __launch_bounds__(256)
conv_weight(const float* __restrict__ W, _Float16* __restrict__ Wh,
            int cout, int cin, int Kp)
{
    int i = blockIdx.x * blockDim.x + threadIdx.x;
    if (i >= cout * Kp) return;
    int c = i / Kp, k = i % Kp;
    Wh[i] = (k < cin) ? (_Float16)W[(size_t)c * cin + k] : (_Float16)0.f;
}

// ---------------------------------------------------------------------------
// Farthest point sampling: one block per batch, LDS-resident distance array.
// ---------------------------------------------------------------------------
__global__ void __launch_bounds__(256)
fps_kernel(const float* __restrict__ xyz, int n, int npoint, int* __restrict__ out)
{
    __shared__ float dist[NPTS];
    __shared__ float rmax[256];
    __shared__ int   rarg[256];
    __shared__ int   sfar;
    const int b = blockIdx.x;
    const float* p = xyz + (size_t)b * n * 3;

    for (int i = threadIdx.x; i < n; i += 256) dist[i] = 1e10f;
    if (threadIdx.x == 0) sfar = 0;
    __syncthreads();

    for (int it = 0; it < npoint; ++it) {
        int far = sfar;
        if (threadIdx.x == 0) out[b * npoint + it] = far;
        float fx = p[far * 3], fy = p[far * 3 + 1], fz = p[far * 3 + 2];
        float lmax = -1.f; int larg = 0;
        for (int i = threadIdx.x; i < n; i += 256) {
            float dx = p[i * 3] - fx, dy = p[i * 3 + 1] - fy, dz = p[i * 3 + 2] - fz;
            float d = fminf(dist[i], dx * dx + dy * dy + dz * dz);
            dist[i] = d;
            if (d > lmax) { lmax = d; larg = i; }
        }
        rmax[threadIdx.x] = lmax; rarg[threadIdx.x] = larg;
        __syncthreads();
        for (int st = 128; st > 0; st >>= 1) {
            if (threadIdx.x < st) {
                float a = rmax[threadIdx.x], c2 = rmax[threadIdx.x + st];
                int ia = rarg[threadIdx.x],  ic = rarg[threadIdx.x + st];
                if (c2 > a || (c2 == a && ic < ia)) { rmax[threadIdx.x] = c2; rarg[threadIdx.x] = ic; }
            }
            __syncthreads();
        }
        if (threadIdx.x == 0) sfar = rarg[0];
        __syncthreads();
    }
}

__global__ void __launch_bounds__(256)
gather_xyz(const float* __restrict__ xyz, const int* __restrict__ fidx,
           float* __restrict__ nxyz, int n, int npoint)
{
    int i = blockIdx.x * blockDim.x + threadIdx.x;
    if (i >= BATCH * npoint) return;
    int b = i / npoint;
    int id = fidx[i];
    const float* s = xyz + ((size_t)b * n + id) * 3;
    nxyz[(size_t)i * 3 + 0] = s[0];
    nxyz[(size_t)i * 3 + 1] = s[1];
    nxyz[(size_t)i * 3 + 2] = s[2];
}

// Ball query: lowest-index nsample points within radius, padded with first hit.
__global__ void __launch_bounds__(256)
ball_query(const float* __restrict__ xyz, const float* __restrict__ nxyz,
           int n, int npoint, float r2, int ns, int* __restrict__ idx)
{
    int t = blockIdx.x * blockDim.x + threadIdx.x;
    if (t >= BATCH * npoint) return;
    int b = t / npoint;
    const float* q = nxyz + (size_t)t * 3;
    const float* p = xyz + (size_t)b * n * 3;
    int cnt = 0, first = 0;
    int* row = idx + (size_t)t * ns;
    for (int j = 0; j < n && cnt < ns; ++j) {
        float dx = p[j * 3] - q[0], dy = p[j * 3 + 1] - q[1], dz = p[j * 3 + 2] - q[2];
        if (dx * dx + dy * dy + dz * dz <= r2) {
            if (cnt == 0) first = j;
            row[cnt++] = j;
        }
    }
    for (int k = cnt; k < ns; ++k) row[k] = first;
}

// Grouping: [relative xyz | gathered features] -> padded f16 GEMM input.
__global__ void __launch_bounds__(256)
group_kernel(const float* __restrict__ xyz, const float* __restrict__ pts,
             const float* __restrict__ nxyz, const int* __restrict__ idx,
             int n, int npoint, int ns, int F, int Kp, _Float16* __restrict__ Xh)
{
    int t = blockIdx.x * blockDim.x + threadIdx.x;
    int total = BATCH * npoint * ns;
    if (t >= total) return;
    int s = (t / ns) % npoint;
    int b = t / (ns * npoint);
    int id = idx[t];
    const float* pp = xyz  + ((size_t)b * n + id) * 3;
    const float* q  = nxyz + ((size_t)b * npoint + s) * 3;
    _Float16* row = Xh + (size_t)t * Kp;
    row[0] = (_Float16)(pp[0] - q[0]);
    row[1] = (_Float16)(pp[1] - q[1]);
    row[2] = (_Float16)(pp[2] - q[2]);
    const float* f = pts + ((size_t)b * n + id) * F;
    for (int c = 0; c < F; ++c) row[3 + c] = (_Float16)f[c];
    for (int c = 3 + F; c < Kp; ++c) row[c] = (_Float16)0.f;
}

// Max-pool over neighbor dimension.
__global__ void __launch_bounds__(256)
maxpool_ns(const float* __restrict__ Y, float* __restrict__ out, int S, int ns, int C)
{
    int i = blockIdx.x * blockDim.x + threadIdx.x;
    if (i >= S * C) return;
    int s = i / C, c = i % C;
    const float* p = Y + (size_t)s * ns * C + c;
    float m = p[0];
    for (int j = 1; j < ns; ++j) m = fmaxf(m, p[(size_t)j * C]);
    out[i] = m;
}

// Feature propagation: 3-NN inverse-distance interpolation + concat -> f16.
__global__ void __launch_bounds__(256)
fp_interp(const float* __restrict__ xyz1, const float* __restrict__ xyz2,
          const float* __restrict__ pts1, const float* __restrict__ pts2,
          int n1, int n2, int C1, int C2, int Kp, _Float16* __restrict__ Xh)
{
    int t = blockIdx.x * blockDim.x + threadIdx.x;
    if (t >= BATCH * n1) return;
    int b = t / n1;
    const float* q  = xyz1 + (size_t)t * 3;
    const float* p2 = xyz2 + (size_t)b * n2 * 3;
    float bd0 = 1e30f, bd1 = 1e30f, bd2 = 1e30f;
    int   bi0 = 0,     bi1 = 0,     bi2 = 0;
    for (int j = 0; j < n2; ++j) {
        float dx = p2[j * 3] - q[0], dy = p2[j * 3 + 1] - q[1], dz = p2[j * 3 + 2] - q[2];
        float d = dx * dx + dy * dy + dz * dz;
        if (d < bd0)      { bd2 = bd1; bi2 = bi1; bd1 = bd0; bi1 = bi0; bd0 = d; bi0 = j; }
        else if (d < bd1) { bd2 = bd1; bi2 = bi1; bd1 = d; bi1 = j; }
        else if (d < bd2) { bd2 = d; bi2 = j; }
    }
    float r0 = 1.f / (fmaxf(bd0, 0.f) + 1e-8f);
    float r1 = 1.f / (fmaxf(bd1, 0.f) + 1e-8f);
    float r2 = 1.f / (fmaxf(bd2, 0.f) + 1e-8f);
    float wsum = r0 + r1 + r2;
    float w0 = r0 / wsum, w1 = r1 / wsum, w2 = r2 / wsum;

    _Float16* row = Xh + (size_t)t * Kp;
    for (int c = 0; c < C1; ++c) row[c] = (_Float16)pts1[(size_t)t * C1 + c];
    const float* g0 = pts2 + ((size_t)b * n2 + bi0) * C2;
    const float* g1 = pts2 + ((size_t)b * n2 + bi1) * C2;
    const float* g2 = pts2 + ((size_t)b * n2 + bi2) * C2;
    for (int c = 0; c < C2; ++c)
        row[C1 + c] = (_Float16)(w0 * g0[c] + w1 * g1[c] + w2 * g2[c]);
    for (int c = C1 + C2; c < Kp; ++c) row[c] = (_Float16)0.f;
}

// Head conv2: 128 -> 1 logit per point.
__global__ void __launch_bounds__(256)
head_conv2(const float* __restrict__ H, const float* __restrict__ w,
           const float* __restrict__ b, float* __restrict__ out, int M)
{
    int m = blockIdx.x * blockDim.x + threadIdx.x;
    if (m >= M) return;
    float acc = b[0];
    const float* h = H + (size_t)m * 128;
    for (int c = 0; c < 128; ++c) acc += h[c] * w[c];
    out[m] = acc;
}

// ---------------------------------------------------------------------------
// Host orchestration
// ---------------------------------------------------------------------------
struct LayerP { const float *W, *b, *beta, *g; int cin, cout; };

static void run_layer(hipStream_t st, const _Float16* Xin, int Kp_in, int M,
                      const LayerP& L, _Float16* Wh, float* Y, _Float16* Xout,
                      float* scale, float* shift)
{
    int wtot = L.cout * Kp_in;
    conv_weight<<<(wtot + 255) / 256, 256, 0, st>>>(L.W, Wh, L.cout, L.cin, Kp_in);

    const int NT = (L.cout == 32) ? 2 : 4;
    dim3 grid(M / 64, L.cout / (16 * NT));
    size_t shbytes = (size_t)NT * 16 * Kp_in * sizeof(_Float16);
    if (NT == 2)
        wmma_gemm_panel<2><<<grid, 128, shbytes, st>>>(Xin, Wh, L.b, Y, M, L.cout, Kp_in);
    else
        wmma_gemm_panel<4><<<grid, 128, shbytes, st>>>(Xin, Wh, L.b, Y, M, L.cout, Kp_in);

    bn_stats<<<L.cout, 256, 0, st>>>(Y, M, L.cout, L.g, L.beta, scale, shift);
    size_t total = (size_t)M * L.cout;
    bn_apply_relu<<<(int)((total + 255) / 256), 256, 0, st>>>(Y, Xout, scale, shift, total, L.cout);
}

static void sa_stage(hipStream_t st, const float* xyz, const float* pts, int n,
                     int npoint, float radius, int ns, int F, const LayerP* L,
                     float* nxyz, float* opts,
                     int* fidx, int* bidx, _Float16* XA, _Float16* XB,
                     _Float16* Wh, float* Y, float* scale, float* shift)
{
    fps_kernel<<<BATCH, 256, 0, st>>>(xyz, n, npoint, fidx);
    int sp = BATCH * npoint;
    gather_xyz<<<(sp + 255) / 256, 256, 0, st>>>(xyz, fidx, nxyz, n, npoint);
    ball_query<<<(sp + 255) / 256, 256, 0, st>>>(xyz, nxyz, n, npoint, radius * radius, ns, bidx);
    int M = sp * ns;
    int Kp = rup(3 + F, 32);
    group_kernel<<<(M + 255) / 256, 256, 0, st>>>(xyz, pts, nxyz, bidx, n, npoint, ns, F, Kp, XA);
    _Float16* bufs[2] = { XA, XB };
    int curb = 0;
    for (int i = 0; i < 3; ++i) {
        _Float16* Xout = (i < 2) ? bufs[1 - curb] : (_Float16*)nullptr;
        run_layer(st, bufs[curb], Kp, M, L[i], Wh, Y, Xout, scale, shift);
        Kp = L[i].cout;
        curb = 1 - curb;
    }
    maxpool_ns<<<(sp * L[2].cout + 255) / 256, 256, 0, st>>>(Y, opts, sp, ns, L[2].cout);
}

// Returns pointer to final-layer f16 activations if produce_f16, else nullptr.
static _Float16* fp_stage(hipStream_t st, const float* xyz1, const float* xyz2,
                          const float* pts1, int C1, const float* pts2, int C2,
                          int n1, int n2, const LayerP* L, int nlayers,
                          float* outpts, bool produce_f16,
                          _Float16* XA, _Float16* XB, _Float16* Wh,
                          float* Y, float* scale, float* shift)
{
    int M = BATCH * n1;
    int Kp = rup(C1 + C2, 32);
    fp_interp<<<(M + 255) / 256, 256, 0, st>>>(xyz1, xyz2, pts1, pts2, n1, n2, C1, C2, Kp, XA);
    _Float16* bufs[2] = { XA, XB };
    int curb = 0;
    _Float16* last = nullptr;
    for (int i = 0; i < nlayers; ++i) {
        bool need_out = (i + 1 < nlayers) || produce_f16;
        _Float16* Xout = need_out ? bufs[1 - curb] : (_Float16*)nullptr;
        run_layer(st, bufs[curb], Kp, M, L[i], Wh, Y, Xout, scale, shift);
        Kp = L[i].cout;
        last = Xout;
        curb = 1 - curb;
    }
    if (outpts)
        (void)hipMemcpyAsync(outpts, Y, (size_t)M * L[nlayers - 1].cout * sizeof(float),
                             hipMemcpyDeviceToDevice, st);
    return last;
}

extern "C" void kernel_launch(void* const* d_in, const int* in_sizes, int n_in,
                              void* d_out, int out_size, void* d_ws, size_t ws_size,
                              hipStream_t stream)
{
    (void)in_sizes; (void)n_in; (void)out_size; (void)ws_size;
    const float* coords   = (const float*)d_in[0];   // [8,4096,3]
    const float* features = (const float*)d_in[1];   // [8,4096,3]

    // Param pytree flatten order: dict keys sorted (fp1,fp2,fp3,fp4,head,sa1..sa4);
    // each layer dict flattens as W,b,beta,g.
    auto LP = [&](int i, int cin, int cout) {
        LayerP l;
        l.W    = (const float*)d_in[i];
        l.b    = (const float*)d_in[i + 1];
        l.beta = (const float*)d_in[i + 2];
        l.g    = (const float*)d_in[i + 3];
        l.cin = cin; l.cout = cout;
        return l;
    };
    LayerP fp1[3] = { LP(2, 128, 128),  LP(6, 128, 128),  LP(10, 128, 128) };
    LayerP fp2[2] = { LP(14, 320, 256), LP(18, 256, 128) };
    LayerP fp3[2] = { LP(22, 384, 256), LP(26, 256, 256) };
    LayerP fp4[2] = { LP(30, 768, 256), LP(34, 256, 256) };
    LayerP headc1 = LP(38, 128, 128);
    const float* w2 = (const float*)d_in[42];   // [1,128]
    const float* b2 = (const float*)d_in[43];   // [1]
    LayerP sa1[3] = { LP(44, 6, 32),    LP(48, 32, 32),   LP(52, 32, 64) };
    LayerP sa2[3] = { LP(56, 67, 64),   LP(60, 64, 64),   LP(64, 64, 128) };
    LayerP sa3[3] = { LP(68, 131, 128), LP(72, 128, 128), LP(76, 128, 256) };
    LayerP sa4[3] = { LP(80, 259, 256), LP(84, 256, 256), LP(88, 256, 512) };

    // Workspace bump allocator (256 B aligned).
    char* base = (char*)d_ws;
    size_t off = 0;
    auto alloc = [&](size_t bytes) -> void* {
        void* p = base + off;
        off = (off + bytes + 255) & ~(size_t)255;
        return p;
    };
    int*   fidx  = (int*)alloc(BATCH * 1024 * sizeof(int));
    int*   bidx  = (int*)alloc((size_t)BATCH * 1024 * 32 * sizeof(int));
    float* l1x   = (float*)alloc((size_t)BATCH * 1024 * 3 * 4);
    float* l2x   = (float*)alloc((size_t)BATCH * 256  * 3 * 4);
    float* l3x   = (float*)alloc((size_t)BATCH * 64   * 3 * 4);
    float* l4x   = (float*)alloc((size_t)BATCH * 16   * 3 * 4);
    float* l1p   = (float*)alloc((size_t)BATCH * 1024 * 64  * 4);
    float* l2p   = (float*)alloc((size_t)BATCH * 256  * 128 * 4);
    float* l3p   = (float*)alloc((size_t)BATCH * 64   * 256 * 4);
    float* l4p   = (float*)alloc((size_t)BATCH * 16   * 512 * 4);
    float* d3    = (float*)alloc((size_t)BATCH * 64   * 256 * 4);  // fp4 out
    float* d2    = (float*)alloc((size_t)BATCH * 256  * 256 * 4);  // fp3 out
    float* d1    = (float*)alloc((size_t)BATCH * 1024 * 128 * 4);  // fp2 out
    _Float16* XA = (_Float16*)alloc((size_t)8388608 * sizeof(_Float16));
    _Float16* XB = (_Float16*)alloc((size_t)8388608 * sizeof(_Float16));
    _Float16* Wh = (_Float16*)alloc((size_t)262144 * sizeof(_Float16));
    float* Y     = (float*)alloc((size_t)16777216 * sizeof(float));
    float* scale = (float*)alloc(512 * sizeof(float));
    float* shift = (float*)alloc(512 * sizeof(float));

    // ---- Set abstraction ----
    sa_stage(stream, coords, features, NPTS, 1024, 0.1f, 32, 3,   sa1, l1x, l1p,
             fidx, bidx, XA, XB, Wh, Y, scale, shift);
    sa_stage(stream, l1x, l1p, 1024, 256, 0.2f, 32, 64,  sa2, l2x, l2p,
             fidx, bidx, XA, XB, Wh, Y, scale, shift);
    sa_stage(stream, l2x, l2p, 256, 64, 0.4f, 32, 128,   sa3, l3x, l3p,
             fidx, bidx, XA, XB, Wh, Y, scale, shift);
    sa_stage(stream, l3x, l3p, 64, 16, 0.8f, 32, 256,    sa4, l4x, l4p,
             fidx, bidx, XA, XB, Wh, Y, scale, shift);

    // ---- Feature propagation ----
    fp_stage(stream, l3x, l4x, l3p, 256, l4p, 512, 64, 16,
             fp4, 2, d3, false, XA, XB, Wh, Y, scale, shift);
    fp_stage(stream, l2x, l3x, l2p, 128, d3, 256, 256, 64,
             fp3, 2, d2, false, XA, XB, Wh, Y, scale, shift);
    fp_stage(stream, l1x, l2x, l1p, 64, d2, 256, 1024, 256,
             fp2, 2, d1, false, XA, XB, Wh, Y, scale, shift);
    _Float16* h16 = fp_stage(stream, coords, l1x, nullptr, 0, d1, 128, NPTS, 1024,
                             fp1, 3, nullptr, true, XA, XB, Wh, Y, scale, shift);

    // ---- Head ----
    int M0 = BATCH * NPTS;
    run_layer(stream, h16, 128, M0, headc1, Wh, Y, nullptr, scale, shift);
    head_conv2<<<(M0 + 255) / 256, 256, 0, stream>>>(Y, w2, b2, (float*)d_out, M0);
}
